// CategoricalGraphAtt_56418690400483
// MI455X (gfx1250) — matrix operand order repaired and checked by reading, not compiled, not run on hardware
//
#include <hip/hip_runtime.h>
#include <hip/hip_bf16.h>
#include <math.h>

typedef __bf16 bf16;
typedef __attribute__((ext_vector_type(16))) __bf16 v16bf;
typedef __attribute__((ext_vector_type(8)))  __bf16 v8bf;
typedef __attribute__((ext_vector_type(8)))  float  v8f;
typedef unsigned int u32x4 __attribute__((ext_vector_type(4)));
typedef int          i32x4 __attribute__((ext_vector_type(4)));
typedef int          i32x8 __attribute__((ext_vector_type(8)));

constexpr int kH  = 1024;
constexpr int kI  = 256;
constexpr int kT  = 128;
constexpr int kNW = 4;
constexpr int kN  = 100;
constexpr int kC  = 5;
constexpr int kG  = 20;
constexpr int kH3 = 3 * kH;

#if defined(__has_builtin)
#if __has_builtin(__builtin_amdgcn_tensor_load_to_lds) && \
    __has_builtin(__builtin_amdgcn_s_wait_tensorcnt)
#define USE_TDM 1
#endif
#endif

// ---------------- conversion / fill ----------------
__global__ void k_cvt_f32_bf16(const float* __restrict__ s, bf16* __restrict__ d, long n) {
  long i = (long)blockIdx.x * blockDim.x + threadIdx.x;
  if (i < n) d[i] = (bf16)s[i];
}
__global__ void k_zero_f32(float* __restrict__ d, long n) {
  long i = (long)blockIdx.x * blockDim.x + threadIdx.x;
  if (i < n) d[i] = 0.0f;
}
__global__ void k_zero_bf16(bf16* __restrict__ d, long n) {
  long i = (long)blockIdx.x * blockDim.x + threadIdx.x;
  if (i < n) d[i] = (bf16)0.0f;
}

// ---- 16-element bf16 fragment from two 16B loads (global or LDS b128) ----
static __device__ __forceinline__ v16bf ld_frag2(const bf16* p0, const bf16* p1) {
  v8bf lo = *(const v8bf*)p0;
  v8bf hi = *(const v8bf*)p1;
  return __builtin_shufflevector(lo, hi, 0, 1, 2, 3, 4, 5, 6, 7,
                                 8, 9, 10, 11, 12, 13, 14, 15);
}

#ifdef USE_TDM
// ---- TDM: async 2D-tile load Global -> LDS (D# per CDNA5 ISA ch.8) ----
// tile_k elems (contiguous, 2B each) x tile_rows rows, row stride = stride elems.
static __device__ __forceinline__ void tdm_load_tile_2d(
    unsigned lds_off, const bf16* gptr, long stride_elems, int tile_k, int tile_rows) {
  unsigned long ga = (unsigned long)(__UINTPTR_TYPE__)gptr;
  u32x4 g0;
  g0.x = 1u;                                                 // count=1, user D#
  g0.y = lds_off;                                            // lds_addr (bytes)
  g0.z = (unsigned)ga;                                       // global_addr lo
  g0.w = (unsigned)((ga >> 32) & 0x1FFFFFFull) | (2u << 30); // addr hi | type=2
  unsigned t0 = (unsigned)tile_k, t1 = (unsigned)tile_rows;
  unsigned long s0 = (unsigned long)stride_elems;
  i32x8 g1;
  g1[0] = 0x10000;                                           // data_size = 2 bytes
  g1[1] = (int)((t0 & 0xFFFFu) << 16);                       // tensor_dim0[15:0]
  g1[2] = (int)(((t0 >> 16) & 0xFFFFu) | ((t1 & 0xFFFFu) << 16)); // t_dim0 hi|t_dim1 lo
  g1[3] = (int)(((t1 >> 16) & 0xFFFFu) | ((t0 & 0xFFFFu) << 16)); // t_dim1 hi|tile_dim0
  g1[4] = (int)(t1 & 0xFFFFu);                               // tile_dim1 (tile_dim2=0)
  g1[5] = (int)(s0 & 0xFFFFFFFFull);                         // tensor_dim0_stride lo
  g1[6] = (int)((s0 >> 32) & 0xFFFFull);                     // stride hi (dim1_stride=0)
  g1[7] = 0;
  i32x4 gz = {0, 0, 0, 0};
#if __clang_major__ >= 23
  i32x8 gz8 = {0, 0, 0, 0, 0, 0, 0, 0};
  __builtin_amdgcn_tensor_load_to_lds(g0, g1, gz, gz, gz8, 0);
#else
  __builtin_amdgcn_tensor_load_to_lds(g0, g1, gz, gz, 0);
#endif
}
#endif  // USE_TDM

// ---------------- WMMA bf16 GEMM:  C(M,N) = A(M,K) * B(N,K)^T + bias, opt relu ----
// Block = 256 threads = 8 waves (4 in M x 2 in N); block tile 64 x 128.
// Requires N % 128 == 0 and K % 64 == 0 (true for all GEMMs here).
// TDM path: B tile (128 rows x 64 K, 16KB) double-buffered in LDS by wave 0,
// completion via s_wait_tensorcnt; A streams direct from global (b128 + prefetch).
__global__ void __launch_bounds__(256) k_wmma_gemm(
    const bf16* __restrict__ A, long lda,
    const bf16* __restrict__ Bw, long ldb,
    const float* __restrict__ bias,
    float* __restrict__ C, long ldc,
    int M, int N, int K, int relu) {
  int lane = threadIdx.x & 31;
  int wave = threadIdx.x >> 5;
  int m0    = blockIdx.y * 64 + (wave >> 1) * 16;   // wave M tile
  int n0loc = (wave & 1) * 64;                      // wave N strip within block
  int n0    = blockIdx.x * 128 + n0loc;

  // A: lane holds row m0+lane%16, K = abase + {0..15}. OOB rows clamp to
  // row 0 (valid memory); their results land in C rows that are never stored.
  int row = m0 + (lane & 15);
  long arow = (long)(row < M ? row : 0) * lda;
  int abase = (lane < 16) ? 0 : 16;
  int bbase = (lane < 16) ? 0 : 16;  // B: lane holds column, K = bbase + {0..15}

  v8f acc0 = {}, acc1 = {}, acc2 = {}, acc3 = {};

#ifdef USE_TDM
  __shared__ bf16 sB[2][128 * 64];                  // 2 x 16KB double buffer
  const bf16* Bblk = Bw + (long)blockIdx.x * 128 * ldb;
  int rl0 = n0loc + 0  + (lane & 15);
  int rl1 = n0loc + 16 + (lane & 15);
  int rl2 = n0loc + 32 + (lane & 15);
  int rl3 = n0loc + 48 + (lane & 15);
  int nchunks = K >> 6;                             // K / 64
  if (wave == 0)
    tdm_load_tile_2d((unsigned)(__UINTPTR_TYPE__)&sB[0][0], Bblk, ldb, 64, 128);
  for (int c = 0; c < nchunks; ++c) {
    if (wave == 0) {
      if (c + 1 < nchunks) {
        tdm_load_tile_2d((unsigned)(__UINTPTR_TYPE__)&sB[(c + 1) & 1][0],
                         Bblk + (long)(c + 1) * 64, ldb, 64, 128);
        __builtin_amdgcn_s_wait_tensorcnt(1);       // chunk c complete
      } else {
        __builtin_amdgcn_s_wait_tensorcnt(0);
      }
    }
    __syncthreads();
    const bf16* sb = &sB[c & 1][0];
    int kg = c * 64;
#pragma unroll
    for (int kk = 0; kk < 64; kk += 32) {
      const bf16* ap = A + arow + kg + kk + abase;
      __builtin_prefetch((const void*)(ap + 32), 0, 1);
      v16bf av = ld_frag2(ap, ap + 8);
      const bf16* b0 = sb + rl0 * 64 + kk + bbase;
      const bf16* b1 = sb + rl1 * 64 + kk + bbase;
      const bf16* b2 = sb + rl2 * 64 + kk + bbase;
      const bf16* b3 = sb + rl3 * 64 + kk + bbase;
      v16bf bv0 = ld_frag2(b0, b0 + 8);
      v16bf bv1 = ld_frag2(b1, b1 + 8);
      v16bf bv2 = ld_frag2(b2, b2 + 8);
      v16bf bv3 = ld_frag2(b3, b3 + 8);
      acc0 = __builtin_amdgcn_wmma_f32_16x16x32_bf16(false, av, false, bv0, (short)0, acc0, false, false);
      acc1 = __builtin_amdgcn_wmma_f32_16x16x32_bf16(false, av, false, bv1, (short)0, acc1, false, false);
      acc2 = __builtin_amdgcn_wmma_f32_16x16x32_bf16(false, av, false, bv2, (short)0, acc2, false, false);
      acc3 = __builtin_amdgcn_wmma_f32_16x16x32_bf16(false, av, false, bv3, (short)0, acc3, false, false);
    }
    __syncthreads();
  }
#else
  long brow0 = (long)(n0 + 0  + (lane & 15)) * ldb;
  long brow1 = (long)(n0 + 16 + (lane & 15)) * ldb;
  long brow2 = (long)(n0 + 32 + (lane & 15)) * ldb;
  long brow3 = (long)(n0 + 48 + (lane & 15)) * ldb;
  for (int k0 = 0; k0 < K; k0 += 32) {
    const bf16* ap = A + arow + k0 + abase;
    __builtin_prefetch((const void*)(ap + 32), 0, 1);
    v16bf av = ld_frag2(ap, ap + 8);
    const bf16* bp0 = Bw + brow0 + k0 + bbase;
    const bf16* bp1 = Bw + brow1 + k0 + bbase;
    const bf16* bp2 = Bw + brow2 + k0 + bbase;
    const bf16* bp3 = Bw + brow3 + k0 + bbase;
    __builtin_prefetch((const void*)(bp0 + 32), 0, 1);
    v16bf bv0 = ld_frag2(bp0, bp0 + 8);
    v16bf bv1 = ld_frag2(bp1, bp1 + 8);
    v16bf bv2 = ld_frag2(bp2, bp2 + 8);
    v16bf bv3 = ld_frag2(bp3, bp3 + 8);
    acc0 = __builtin_amdgcn_wmma_f32_16x16x32_bf16(false, av, false, bv0, (short)0, acc0, false, false);
    acc1 = __builtin_amdgcn_wmma_f32_16x16x32_bf16(false, av, false, bv1, (short)0, acc1, false, false);
    acc2 = __builtin_amdgcn_wmma_f32_16x16x32_bf16(false, av, false, bv2, (short)0, acc2, false, false);
    acc3 = __builtin_amdgcn_wmma_f32_16x16x32_bf16(false, av, false, bv3, (short)0, acc3, false, false);
  }
#endif

  // C: VGPR r -> M = r + 8*(lane>=16), N = lane%16 (+ tile offset)
  int rofs = (lane < 16) ? 0 : 8;
  int cbase = n0 + (lane & 15);
  v8f* accs[4] = {&acc0, &acc1, &acc2, &acc3};
#pragma unroll
  for (int j = 0; j < 4; ++j) {
    int ccol = cbase + j * 16;
    float badd = bias ? bias[ccol] : 0.0f;
#pragma unroll
    for (int r = 0; r < 8; ++r) {
      int crow = m0 + r + rofs;
      if (crow < M) {
        float v = (*accs[j])[r] + badd;
        if (relu) v = fmaxf(v, 0.0f);
        C[(long)crow * ldc + ccol] = v;
      }
    }
  }
}

// ---------------- GRU gate fusion: gx,gh (M,3H) -> h (M,H), seq slot -------------
__global__ void k_gru_gate(const float* __restrict__ gx, const float* __restrict__ gh,
                           float* __restrict__ h, bf16* __restrict__ h_bf,
                           bf16* __restrict__ seq, long seq_row_stride, int M) {
  long i = (long)blockIdx.x * blockDim.x + threadIdx.x;
  if (i >= (long)M * kH) return;
  int m = (int)(i / kH), c = (int)(i % kH);
  long g = (long)m * kH3;
  float xr = gx[g + c], xz = gx[g + kH + c], xn = gx[g + 2 * kH + c];
  float hr = gh[g + c], hz = gh[g + kH + c], hn = gh[g + 2 * kH + c];
  float r  = 1.0f / (1.0f + __expf(-(xr + hr)));
  float z  = 1.0f / (1.0f + __expf(-(xz + hz)));
  float nn = tanhf(xn + r * hn);
  float hv = (1.0f - z) * nn + z * h[i];
  h[i]    = hv;
  h_bf[i] = (bf16)hv;
  seq[(long)m * seq_row_stride + c] = (bf16)hv;
}

// ---------------- AttentionBlock: x (B,T,D) bf16, W (T,T), b (T) -> out (B,D) ----
__global__ void k_att(const bf16* __restrict__ x, const float* __restrict__ W,
                      const float* __restrict__ bv, float* __restrict__ out,
                      int T, int D, long sb, long st, long so) {
  __shared__ float lg[64][129];
  int b = blockIdx.x;
  int d = blockIdx.y * 64 + threadIdx.x;
  if (d >= D) return;
  const bf16* xb = x + (long)b * sb + d;
  float m = -3.0e38f;
  for (int tp = 0; tp < T; ++tp) {
    float s = bv[tp];
    for (int t = 0; t < T; ++t) s += (float)xb[(long)t * st] * W[tp * T + t];
    lg[threadIdx.x][tp] = s;
    m = fmaxf(m, s);
  }
  float sum = 0.0f;
  for (int tp = 0; tp < T; ++tp) {
    float e = __expf(lg[threadIdx.x][tp] - m);
    lg[threadIdx.x][tp] = e;
    sum += e;
  }
  float acc = 0.0f;
  for (int tp = 0; tp < T; ++tp) acc += lg[threadIdx.x][tp] * (float)xb[(long)tp * st];
  out[(long)b * so + d] = acc / sum;
}

// ---------------- GAT helpers ----------------------------------------------------
__global__ void k_dots(const float* __restrict__ h, const float* __restrict__ a1,
                       const float* __restrict__ a2, float* __restrict__ o1,
                       float* __restrict__ o2, int Hd) {
  __shared__ float r1[256], r2[256];
  int i = blockIdx.x;
  float s1 = 0.0f, s2 = 0.0f;
  for (int c = threadIdx.x; c < Hd; c += 256) {
    float v = h[(long)i * Hd + c];
    s1 += v * a1[c];
    s2 += v * a2[c];
  }
  r1[threadIdx.x] = s1; r2[threadIdx.x] = s2;
  __syncthreads();
  for (int o = 128; o > 0; o >>= 1) {
    if (threadIdx.x < o) { r1[threadIdx.x] += r1[threadIdx.x + o]; r2[threadIdx.x] += r2[threadIdx.x + o]; }
    __syncthreads();
  }
  if (threadIdx.x == 0) { o1[i] = r1[0]; o2[i] = r2[0]; }
}

__global__ void k_init_bias(float* __restrict__ out, const float* __restrict__ bias,
                            int n, int Hd) {
  long i = (long)blockIdx.x * blockDim.x + threadIdx.x;
  if (i < (long)n * Hd) out[i] = bias[(int)(i % Hd)];
}

__global__ void k_gat_alpha(const int* __restrict__ edges, int E0, int n,
                            const float* __restrict__ as_, const float* __restrict__ ad_,
                            float* __restrict__ alpha, int* __restrict__ sd,
                            float* __restrict__ mbuf, float* __restrict__ sbuf) {
  if (threadIdx.x != 0 || blockIdx.x != 0) return;
  int E = E0 + n;
  for (int i = 0; i < n; ++i) { mbuf[i] = -3.0e38f; sbuf[i] = 0.0f; }
  for (int e = 0; e < E; ++e) {
    int s   = (e < E0) ? edges[e]      : (e - E0);
    int dst = (e < E0) ? edges[E0 + e] : (e - E0);
    float v = as_[s] + ad_[dst];
    v = (v > 0.0f) ? v : 0.2f * v;                 // leaky_relu 0.2
    alpha[e] = v; sd[2 * e] = s; sd[2 * e + 1] = dst;
    if (v > mbuf[dst]) mbuf[dst] = v;
  }
  for (int e = 0; e < E; ++e) {
    int dst = sd[2 * e + 1];
    float ex = __expf(alpha[e] - mbuf[dst]);
    alpha[e] = ex; sbuf[dst] += ex;
  }
  for (int e = 0; e < E; ++e) alpha[e] /= sbuf[sd[2 * e + 1]];
}

__global__ void k_gat_scatter(const float* __restrict__ hsrc, const float* __restrict__ alpha,
                              const int* __restrict__ sd, float* __restrict__ out, int Hd) {
  int e = blockIdx.x;
  int c = blockIdx.y * blockDim.x + threadIdx.x;
  if (c >= Hd) return;
  atomicAdd(&out[(long)sd[2 * e + 1] * Hd + c], alpha[e] * hsrc[(long)sd[2 * e] * Hd + c]);
}

// ---------------- concat (wav | cat_broadcast | inner) -> bf16 (100,3072) --------
__global__ void k_concat(const float* __restrict__ wav, const float* __restrict__ catg,
                         const float* __restrict__ inner, bf16* __restrict__ out) {
  long i = (long)blockIdx.x * blockDim.x + threadIdx.x;
  if (i >= (long)kN * kH3) return;
  int n = (int)(i / kH3), c = (int)(i % kH3);
  float v;
  if (c < kH)          v = wav[(long)n * kH + c];
  else if (c < 2 * kH) v = catg[(long)(n / kG) * kH + (c - kH)];
  else                 v = inner[(long)n * kH + (c - 2 * kH)];
  out[i] = (bf16)v;
}

// ---------------- heads: reg + sigmoid(cls) --------------------------------------
__global__ void k_heads(const float* __restrict__ f, const float* __restrict__ regW,
                        const float* __restrict__ regb, const float* __restrict__ clsW,
                        const float* __restrict__ clsb, float* __restrict__ out) {
  __shared__ float r1[256], r2[256];
  int row = blockIdx.x;
  float s1 = 0.0f, s2 = 0.0f;
  for (int c = threadIdx.x; c < kH; c += 256) {
    float v = f[(long)row * kH + c];
    s1 += v * regW[c];
    s2 += v * clsW[c];
  }
  r1[threadIdx.x] = s1; r2[threadIdx.x] = s2;
  __syncthreads();
  for (int o = 128; o > 0; o >>= 1) {
    if (threadIdx.x < o) { r1[threadIdx.x] += r1[threadIdx.x + o]; r2[threadIdx.x] += r2[threadIdx.x + o]; }
    __syncthreads();
  }
  if (threadIdx.x == 0) {
    out[row]      = r1[0] + regb[0];
    out[kN + row] = 1.0f / (1.0f + __expf(-(r2[0] + clsb[0])));
  }
}

// =================================================================================
extern "C" void kernel_launch(void* const* d_in, const int* in_sizes, int n_in,
                              void* d_out, int out_size, void* d_ws, size_t ws_size,
                              hipStream_t stream) {
  (void)in_sizes; (void)n_in; (void)out_size; (void)ws_size;

  const float* w[4] = {(const float*)d_in[0], (const float*)d_in[1],
                       (const float*)d_in[2], (const float*)d_in[3]};
  const int*   inner_edge = (const int*)d_in[4];
  const int*   outer_edge = (const int*)d_in[5];
  const float* enc_Wih = (const float*)d_in[6];
  const float* enc_Whh = (const float*)d_in[7];
  const float* enc_bih = (const float*)d_in[8];
  const float* enc_bhh = (const float*)d_in[9];
  const float* enc_attW = (const float*)d_in[10];
  const float* enc_attb = (const float*)d_in[11];
  const float* wk_Wih = (const float*)d_in[12];
  const float* wk_Whh = (const float*)d_in[13];
  const float* wk_bih = (const float*)d_in[14];
  const float* wk_bhh = (const float*)d_in[15];
  const float* wkatt_W = (const float*)d_in[16];
  const float* wkatt_b = (const float*)d_in[17];
  const float* pool_W = (const float*)d_in[18];
  const float* pool_b = (const float*)d_in[19];
  const float* ig_W  = (const float*)d_in[20];
  const float* ig_as = (const float*)d_in[21];
  const float* ig_ad = (const float*)d_in[22];
  const float* ig_b  = (const float*)d_in[23];
  const float* cg_W  = (const float*)d_in[24];
  const float* cg_as = (const float*)d_in[25];
  const float* cg_ad = (const float*)d_in[26];
  const float* cg_b  = (const float*)d_in[27];
  const float* fus_W = (const float*)d_in[28];
  const float* fus_b = (const float*)d_in[29];
  const float* reg_W = (const float*)d_in[30];
  const float* reg_b = (const float*)d_in[31];
  const float* cls_W = (const float*)d_in[32];
  const float* cls_b = (const float*)d_in[33];
  float* out = (float*)d_out;

  // ---- workspace bump allocator ----
  size_t off = 0;
  auto alloc = [&](size_t bytes) -> void* {
    off = (off + 255) & ~(size_t)255;
    void* r = (char*)d_ws + off;
    off += bytes;
    return r;
  };

  bf16* wbf       = (bf16*)alloc((size_t)4 * kN * kT * kI * 2);
  bf16* encWih_bf = (bf16*)alloc((size_t)4 * kH3 * kI * 2);
  bf16* encWhh_bf = (bf16*)alloc((size_t)4 * kH3 * kH * 2);
  bf16* wkWih_bf  = (bf16*)alloc((size_t)kH3 * kH * 2);
  bf16* wkWhh_bf  = (bf16*)alloc((size_t)kH3 * kH * 2);
  bf16* igW_bf    = (bf16*)alloc((size_t)kH * kH * 2);
  bf16* cgW_bf    = (bf16*)alloc((size_t)kH * kH * 2);
  bf16* fusW_bf   = (bf16*)alloc((size_t)kH * kH3 * 2);
  bf16* seq_bf    = (bf16*)alloc((size_t)kN * kT * kH * 2);
  float* gx       = (float*)alloc((size_t)kN * kH3 * 4);
  float* gh       = (float*)alloc((size_t)kN * kH3 * 4);
  float* h        = (float*)alloc((size_t)kN * kH * 4);
  bf16*  h_bf     = (bf16*)alloc((size_t)kN * kH * 2);
  float* we_f     = (float*)alloc((size_t)kN * kNW * kH * 4);
  bf16*  we_bf    = (bf16*)alloc((size_t)kN * kNW * kH * 2);
  bf16*  we2_bf   = (bf16*)alloc((size_t)kN * kNW * kH * 2);
  float* wav      = (float*)alloc((size_t)kN * kH * 4);
  bf16*  wav_bf   = (bf16*)alloc((size_t)kN * kH * 2);
  float* hin      = (float*)alloc((size_t)kN * kH * 4);
  float* inner    = (float*)alloc((size_t)kN * kH * 4);
  float* catp     = (float*)alloc((size_t)kC * kH * 4);
  bf16*  catp_bf  = (bf16*)alloc((size_t)kC * kH * 2);
  float* hout     = (float*)alloc((size_t)kC * kH * 4);
  float* catg     = (float*)alloc((size_t)kC * kH * 4);
  bf16*  cc_bf    = (bf16*)alloc((size_t)kN * kH3 * 2);
  float* fbuf     = (float*)alloc((size_t)kN * kH * 4);
  float* asbuf    = (float*)alloc(128 * 4);
  float* adbuf    = (float*)alloc(128 * 4);
  float* mbuf     = (float*)alloc(128 * 4);
  float* sbuf     = (float*)alloc(128 * 4);
  float* alpha    = (float*)alloc(2112 * 4);
  int*   sd       = (int*)alloc(2 * 2112 * 4);

  auto cvt = [&](const float* s, bf16* d, long n) {
    k_cvt_f32_bf16<<<dim3((unsigned)((n + 255) / 256)), dim3(256), 0, stream>>>(s, d, n);
  };
  auto zeroF = [&](float* d, long n) {
    k_zero_f32<<<dim3((unsigned)((n + 255) / 256)), dim3(256), 0, stream>>>(d, n);
  };
  auto zeroB = [&](bf16* d, long n) {
    k_zero_bf16<<<dim3((unsigned)((n + 255) / 256)), dim3(256), 0, stream>>>(d, n);
  };
  auto gemm = [&](const bf16* A, long lda, const bf16* Bw, long ldb, const float* bias,
                  float* Cm, long ldc, int M, int Nn, int K, int relu) {
    dim3 g((unsigned)((Nn + 127) / 128), (unsigned)((M + 63) / 64));
    k_wmma_gemm<<<g, dim3(256), 0, stream>>>(A, lda, Bw, ldb, bias, Cm, ldc, M, Nn, K, relu);
  };

  // ---- one-time fp32 -> bf16 conversions ----
  for (int i = 0; i < 4; ++i) cvt(w[i], wbf + (long)i * kN * kT * kI, (long)kN * kT * kI);
  cvt(enc_Wih, encWih_bf, (long)4 * kH3 * kI);
  cvt(enc_Whh, encWhh_bf, (long)4 * kH3 * kH);
  cvt(wk_Wih, wkWih_bf, (long)kH3 * kH);
  cvt(wk_Whh, wkWhh_bf, (long)kH3 * kH);
  cvt(ig_W, igW_bf, (long)kH * kH);
  cvt(cg_W, cgW_bf, (long)kH * kH);
  cvt(fus_W, fusW_bf, (long)kH * kH3);

  // ---- per-week GRU encoders + time attention ----
  for (int i = 0; i < kNW; ++i) {
    zeroF(h, (long)kN * kH);
    zeroB(h_bf, (long)kN * kH);
    const bf16* wbf_i = wbf + (long)i * kN * kT * kI;
    const bf16* Wih_i = encWih_bf + (long)i * kH3 * kI;
    const bf16* Whh_i = encWhh_bf + (long)i * kH3 * kH;
    const float* bih_i = enc_bih + (long)i * kH3;
    const float* bhh_i = enc_bhh + (long)i * kH3;
    for (int t = 0; t < kT; ++t) {
      gemm(wbf_i + (long)t * kI, (long)kT * kI, Wih_i, kI, bih_i, gx, kH3, kN, kH3, kI, 0);
      gemm(h_bf, kH, Whh_i, kH, bhh_i, gh, kH3, kN, kH3, kH, 0);
      k_gru_gate<<<dim3((kN * kH + 255) / 256), dim3(256), 0, stream>>>(
          gx, gh, h, h_bf, seq_bf + (long)t * kH, (long)kT * kH, kN);
    }
    k_att<<<dim3(kN, kH / 64), dim3(64), 0, stream>>>(
        seq_bf, enc_attW + (long)i * kT * kT, enc_attb + (long)i * kT,
        we_f + (long)i * kH, kT, kH, (long)kT * kH, (long)kH, (long)kNW * kH);
  }

  // ---- weekly GRU over stocks (seq len 100, batch 4) + week attention ----
  cvt(we_f, we_bf, (long)kN * kNW * kH);
  zeroF(h, (long)kNW * kH);
  zeroB(h_bf, (long)kNW * kH);
  for (int t = 0; t < kN; ++t) {
    gemm(we_bf + (long)t * kNW * kH, kH, wkWih_bf, kH, wk_bih, gx, kH3, kNW, kH3, kH, 0);
    gemm(h_bf, kH, wkWhh_bf, kH, wk_bhh, gh, kH3, kNW, kH3, kH, 0);
    k_gru_gate<<<dim3((kNW * kH + 255) / 256), dim3(256), 0, stream>>>(
        gx, gh, h, h_bf, we2_bf + (long)t * kNW * kH, (long)kH, kNW);
  }
  k_att<<<dim3(kN, kH / 64), dim3(64), 0, stream>>>(
      we2_bf, wkatt_W, wkatt_b, wav, kNW, kH, (long)kNW * kH, (long)kH, (long)kH);

  // ---- inner GAT on wav (100 nodes) ----
  cvt(wav, wav_bf, (long)kN * kH);
  gemm(wav_bf, kH, igW_bf, kH, nullptr, hin, kH, kN, kH, kH, 0);
  k_dots<<<dim3(kN), dim3(256), 0, stream>>>(hin, ig_as, ig_ad, asbuf, adbuf, kH);
  k_init_bias<<<dim3((kN * kH + 255) / 256), dim3(256), 0, stream>>>(inner, ig_b, kN, kH);
  k_gat_alpha<<<dim3(1), dim3(64), 0, stream>>>(inner_edge, 2000, kN, asbuf, adbuf, alpha, sd, mbuf, sbuf);
  k_gat_scatter<<<dim3(2000 + kN, kH / 256), dim3(256), 0, stream>>>(hin, alpha, sd, inner, kH);

  // ---- category pooling attention (C=5, G=20) ----
  k_att<<<dim3(kC, kH / 64), dim3(64), 0, stream>>>(
      wav_bf, pool_W, pool_b, catp, kG, kH, (long)kG * kH, (long)kH, (long)kH);

  // ---- outer GAT on categories (5 nodes) ----
  cvt(catp, catp_bf, (long)kC * kH);
  gemm(catp_bf, kH, cgW_bf, kH, nullptr, hout, kH, kC, kH, kH, 0);
  k_dots<<<dim3(kC), dim3(256), 0, stream>>>(hout, cg_as, cg_ad, asbuf, adbuf, kH);
  k_init_bias<<<dim3((kC * kH + 255) / 256), dim3(256), 0, stream>>>(catg, cg_b, kC, kH);
  k_gat_alpha<<<dim3(1), dim3(64), 0, stream>>>(outer_edge, 25, kC, asbuf, adbuf, alpha, sd, mbuf, sbuf);
  k_gat_scatter<<<dim3(25 + kC, kH / 256), dim3(256), 0, stream>>>(hout, alpha, sd, catg, kH);

  // ---- fusion + heads ----
  k_concat<<<dim3((unsigned)(((long)kN * kH3 + 255) / 256)), dim3(256), 0, stream>>>(
      wav, catg, inner, cc_bf);
  gemm(cc_bf, kH3, fusW_bf, kH3, fus_b, fbuf, kH, kN, kH, kH3, 1);
  k_heads<<<dim3(kN), dim3(256), 0, stream>>>(fbuf, reg_W, reg_b, cls_W, cls_b, out);
}
